// SpotGuidedTransformerS2_77214922047579
// MI455X (gfx1250) — matrix operand order repaired and checked by reading, not compiled, not run on hardware
//
#include <hip/hip_runtime.h>
#include <math.h>

// ---------------- problem constants (from reference) ----------------
#define BN_     2048   // N == M
#define CIN_    1024
#define D_      256
#define H_      4
#define DH_     64
#define KNN_    17     // kmax = max(K+1, SPOT_K) = 17
#define SPOTK_  12
#define SPOTS_  8
#define NBLOCKS_ 3
#define SEED_   128
#define DFF_    512
#define THR_    0.5f
#define SIGC_   1.0f
#define SCALE_  0.125f // 1/sqrt(DH)

typedef __bf16 bf16;
typedef __attribute__((ext_vector_type(16))) __bf16 v16bf;
typedef __attribute__((ext_vector_type(8)))  __bf16 v8bf;
typedef __attribute__((ext_vector_type(8)))  float  v8f;

// ---------------- squared pairwise distances ----------------
__global__ void k_pdist(const float* __restrict__ x, const float* __restrict__ y,
                        float* __restrict__ d, int n, int m) {
  int j = blockIdx.x * blockDim.x + threadIdx.x;
  int i = blockIdx.y;
  if (j >= m) return;
  float ax = x[i*3+0], ay = x[i*3+1], az = x[i*3+2];
  float bx = y[j*3+0], by = y[j*3+1], bz = y[j*3+2];
  float dx = ax-bx, dy = ay-by, dz = az-bz;
  d[(long)i*m + j] = dx*dx + dy*dy + dz*dz;
}

// ---------------- per-row 17 nearest (smallest distance) ----------------
__global__ void k_knn(const float* __restrict__ d, int n, int* __restrict__ idx) {
  int row = blockIdx.x * blockDim.x + threadIdx.x;
  if (row >= n) return;
  float bd[KNN_]; int bix[KNN_];
  for (int t = 0; t < KNN_; ++t) { bd[t] = 3.0e38f; bix[t] = 0; }
  const float* dr = d + (long)row*n;
  for (int c = 0; c < n; ++c) {
    float v = dr[c];
    if (v < bd[KNN_-1]) {
      int p = KNN_-1;
      while (p > 0 && bd[p-1] > v) { bd[p] = bd[p-1]; bix[p] = bix[p-1]; --p; }
      bd[p] = v; bix[p] = c;
    }
  }
  for (int t = 0; t < KNN_; ++t) idx[row*KNN_ + t] = bix[t];
}

// ---------------- generic bf16-WMMA GEMM ----------------
// C[rows x cols] = act(A[rows x kdim] * B (+bias)) (+resid)
// B is kdim x cols row-major if tB==0; cols x kdim row-major if tB==1 (B^T).
// WG = 128 threads = 4 waves; WG tile 128x64; wave tile 32x64 (8 WMMAs/K-step).
// LDS tiles are stored in WMMA fragment-major layout: each lane's 16 bf16
// fragment values are contiguous (2 x ds_load_b128 per fragment).
#define LSTRIDE_ 24  // bf16 per lane slot (48B: 16B-aligned, bank-friendly)
__global__ __launch_bounds__(128) void k_gemm(
    const float* __restrict__ A, int lda,
    const float* __restrict__ B, int ldb, int tB,
    const float* __restrict__ bias, const float* __restrict__ resid,
    float* __restrict__ C, int ldc, int kdim, int relu)
{
  __shared__ __attribute__((aligned(64))) bf16 AsF[8][32][LSTRIDE_]; // [m-sub][lane][elem]
  __shared__ __attribute__((aligned(64))) bf16 BsF[4][32][LSTRIDE_]; // [n-sub][lane][elem]
  int t = threadIdx.x;
  int w = t >> 5, lane = t & 31;
  int brow = blockIdx.y * 128, bcol = blockIdx.x * 64;
  int mrow = lane & 15, khalf = lane >> 4;

  v8f acc[2][4] = {};

  union FU { v16bf v; v8bf h[2]; };

  for (int k0 = 0; k0 < kdim; k0 += 32) {
    if (k0 + 32 < kdim)
      __builtin_prefetch(&A[(long)(brow + t) * lda + (k0 + 32)], 0, 1);
    // ---- stage A: 128x32 in octets; 512 chunks, 4 per thread
#pragma unroll
    for (int ch = 0; ch < 4; ++ch) {
      int id = t + ch * 128;          // 0..511
      int r  = id >> 2;               // 0..127
      int c0 = (id & 3) * 8;          // 0,8,16,24
      const float* ap = &A[(long)(brow + r) * lda + (k0 + c0)];
      float4 f0 = *(const float4*)ap;
      float4 f1 = *(const float4*)(ap + 4);
      v8bf pk;
      pk[0] = (bf16)f0.x; pk[1] = (bf16)f0.y; pk[2] = (bf16)f0.z; pk[3] = (bf16)f0.w;
      pk[4] = (bf16)f1.x; pk[5] = (bf16)f1.y; pk[6] = (bf16)f1.z; pk[7] = (bf16)f1.w;
      int kh = (c0 >> 3) & 1;         // K-half owning lane group
      int e0 = (c0 & 16) ? 8 : 0;     // element offset within fragment
      *(v8bf*)&AsF[r >> 4][kh*16 + (r & 15)][e0] = pk;
    }
    // ---- stage B: 32x64 in K-octets; 256 chunks, 2 per thread
#pragma unroll
    for (int ch = 0; ch < 2; ++ch) {
      int id = t + ch * 128;          // 0..255
      int c  = id >> 2;               // 0..63
      int r0 = (id & 3) * 8;          // 0,8,16,24
      v8bf pk;
      if (tB) {
        const float* bp = &B[(long)(bcol + c) * ldb + (k0 + r0)];
        float4 f0 = *(const float4*)bp;
        float4 f1 = *(const float4*)(bp + 4);
        pk[0] = (bf16)f0.x; pk[1] = (bf16)f0.y; pk[2] = (bf16)f0.z; pk[3] = (bf16)f0.w;
        pk[4] = (bf16)f1.x; pk[5] = (bf16)f1.y; pk[6] = (bf16)f1.z; pk[7] = (bf16)f1.w;
      } else {
#pragma unroll
        for (int u = 0; u < 8; ++u)
          pk[u] = (bf16)B[(long)(k0 + r0 + u) * ldb + (bcol + c)];
      }
      *(v8bf*)&BsF[c >> 4][((r0 >> 3) & 1)*16 + (c & 15)][(r0 >> 1) & 8] = pk;
    }
    __syncthreads();

    FU a0, a1;
    a0.h[0] = *(const v8bf*)&AsF[2*w + 0][lane][0];
    a0.h[1] = *(const v8bf*)&AsF[2*w + 0][lane][8];
    a1.h[0] = *(const v8bf*)&AsF[2*w + 1][lane][0];
    a1.h[1] = *(const v8bf*)&AsF[2*w + 1][lane][8];
#pragma unroll
    for (int nt = 0; nt < 4; ++nt) {
      FU bb;
      bb.h[0] = *(const v8bf*)&BsF[nt][lane][0];
      bb.h[1] = *(const v8bf*)&BsF[nt][lane][8];
      acc[0][nt] = __builtin_amdgcn_wmma_f32_16x16x32_bf16(
          false, a0.v, false, bb.v, (short)0, acc[0][nt], false, false);
      acc[1][nt] = __builtin_amdgcn_wmma_f32_16x16x32_bf16(
          false, a1.v, false, bb.v, (short)0, acc[1][nt], false, false);
    }
    __syncthreads();
  }

  // epilogue: C/D layout lanes 0-15: M=v, N=lane; lanes 16-31: M=v+8, N=lane-16
#pragma unroll
  for (int mh = 0; mh < 2; ++mh) {
#pragma unroll
    for (int nt = 0; nt < 4; ++nt) {
#pragma unroll
      for (int v = 0; v < 8; ++v) {
        int gr = brow + (2*w + mh)*16 + v + khalf*8;
        int gc = bcol + nt*16 + mrow;
        float val = acc[mh][nt][v];
        if (bias)  val += bias[gc];
        if (relu)  val = fmaxf(val, 0.0f);
        if (resid) val += resid[(long)gr*ldc + gc];
        C[(long)gr*ldc + gc] = val;
      }
    }
  }
}

// ---------------- row softmax stats ----------------
__global__ void k_rowstats(const float* __restrict__ S, int n,
                           float* __restrict__ rmax, float* __restrict__ rsum) {
  int r = blockIdx.x, t = threadIdx.x;
  __shared__ float red[256];
  const float* sr = S + (long)r*n;
  float lm = -3.0e38f;
  for (int c = t; c < n; c += 256) lm = fmaxf(lm, sr[c]);
  red[t] = lm; __syncthreads();
  for (int s = 128; s > 0; s >>= 1) { if (t < s) red[t] = fmaxf(red[t], red[t+s]); __syncthreads(); }
  float mx = red[0]; __syncthreads();
  float ls = 0.0f;
  for (int c = t; c < n; c += 256) ls += __expf(sr[c] - mx);
  red[t] = ls; __syncthreads();
  for (int s = 128; s > 0; s >>= 1) { if (t < s) red[t] += red[t+s]; __syncthreads(); }
  if (!t) { rmax[r] = mx; rsum[r] = red[0]; }
}

__global__ void k_colstats(const float* __restrict__ S, int n,
                           float* __restrict__ cmax, float* __restrict__ csum) {
  int c = blockIdx.x, t = threadIdx.x;
  __shared__ float red[256];
  float lm = -3.0e38f;
  for (int r = t; r < n; r += 256) lm = fmaxf(lm, S[(long)r*n + c]);
  red[t] = lm; __syncthreads();
  for (int s = 128; s > 0; s >>= 1) { if (t < s) red[t] = fmaxf(red[t], red[t+s]); __syncthreads(); }
  float mx = red[0]; __syncthreads();
  float ls = 0.0f;
  for (int r = t; r < n; r += 256) ls += __expf(S[(long)r*n + c] - mx);
  red[t] = ls; __syncthreads();
  for (int s = 128; s > 0; s >>= 1) { if (t < s) red[t] += red[t+s]; __syncthreads(); }
  if (!t) { cmax[c] = mx; csum[c] = red[0]; }
}

// ---------------- dual-normalized matching matrix (float4 per thread) ----------------
__global__ void k_dualsoftmax(const float* __restrict__ S,
                              const float* __restrict__ rmax, const float* __restrict__ rsum,
                              const float* __restrict__ cmax, const float* __restrict__ csum,
                              float* __restrict__ ms, int n) {
  long t = ((long)blockIdx.x * blockDim.x + threadIdx.x) * 4;
  if (t >= (long)n*n) return;
  int r = (int)(t / n), c = (int)(t % n);
  float4 s4 = *(const float4*)(S + t);
  float rm = rmax[r], rsi = 1.0f / rsum[r];
  float4 o;
  o.x = __expf(s4.x - rm) * rsi * (__expf(s4.x - cmax[c+0]) / csum[c+0]);
  o.y = __expf(s4.y - rm) * rsi * (__expf(s4.y - cmax[c+1]) / csum[c+1]);
  o.z = __expf(s4.z - rm) * rsi * (__expf(s4.z - cmax[c+2]) / csum[c+2]);
  o.w = __expf(s4.w - rm) * rsi * (__expf(s4.w - cmax[c+3]) / csum[c+3]);
  *(float4*)(ms + t) = o;
}

// ---------------- row / col argmax (conf, mi) ----------------
__global__ void k_rowargmax(const float* __restrict__ ms, int n,
                            float* __restrict__ conf, int* __restrict__ mi) {
  int r = blockIdx.x, t = threadIdx.x;
  __shared__ float sv[256]; __shared__ int si[256];
  float bv = -3.0e38f; int bi2 = n;
  const float* mr = ms + (long)r*n;
  for (int c = t; c < n; c += 256) {
    float v = mr[c];
    if (v > bv || (v == bv && c < bi2)) { bv = v; bi2 = c; }
  }
  sv[t] = bv; si[t] = bi2; __syncthreads();
  for (int s = 128; s > 0; s >>= 1) {
    if (t < s) {
      if (sv[t+s] > sv[t] || (sv[t+s] == sv[t] && si[t+s] < si[t])) { sv[t] = sv[t+s]; si[t] = si[t+s]; }
    }
    __syncthreads();
  }
  if (!t) { conf[r] = sv[0]; mi[r] = si[0]; }
}

__global__ void k_colargmax(const float* __restrict__ ms, int n,
                            float* __restrict__ conf, int* __restrict__ mi) {
  int c = blockIdx.x, t = threadIdx.x;
  __shared__ float sv[256]; __shared__ int si[256];
  float bv = -3.0e38f; int bi2 = n;
  for (int r = t; r < n; r += 256) {
    float v = ms[(long)r*n + c];
    if (v > bv || (v == bv && r < bi2)) { bv = v; bi2 = r; }
  }
  sv[t] = bv; si[t] = bi2; __syncthreads();
  for (int s = 128; s > 0; s >>= 1) {
    if (t < s) {
      if (sv[t+s] > sv[t] || (sv[t+s] == sv[t] && si[t+s] < si[t])) { sv[t] = sv[t+s]; si[t] = si[t+s]; }
    }
    __syncthreads();
  }
  if (!t) { conf[c] = sv[0]; mi[c] = si[0]; }
}

// ---------------- spatial compatibility ----------------
__global__ void k_compat(const float* __restrict__ dself, const float* __restrict__ dcross,
                         const int* __restrict__ mi, const float* __restrict__ conf,
                         float* __restrict__ comp, float* __restrict__ c2,
                         float* __restrict__ outSlice, int bi, int n) {
  int i = blockIdx.x, t = threadIdx.x;
  __shared__ float red[256];
  const float* drow = dself + (long)i*n;
  const float* grow = dcross + (long)mi[i]*n;
  float acc = 0.0f;
  for (int j = t; j < n; j += 256) {
    float g = grow[mi[j]];
    float c = 1.0f - fabsf(drow[j] - g) / SIGC_;
    acc += fmaxf(c, 0.0f);
  }
  red[t] = acc; __syncthreads();
  for (int s = 128; s > 0; s >>= 1) { if (t < s) red[t] += red[t+s]; __syncthreads(); }
  if (!t) {
    float cm = red[0] / (float)n;
    comp[i] = cm;
    c2[i] = conf[i] * cm;
    outSlice[i*NBLOCKS_ + bi] = cm;
  }
}

// ---------------- seeding: iterative top-128 (single workgroup) ----------------
__global__ void k_seeding(const float* __restrict__ comp, const float* __restrict__ c2,
                          int* __restrict__ sidx, int* __restrict__ smask, int n) {
  int t = threadIdx.x;
  __shared__ float sc[BN_];
  __shared__ float sv[256]; __shared__ int si[256];
  __shared__ float s_max; __shared__ int s_cnt;
  float lm = -3.0e38f;
  for (int i = t; i < n; i += 256) lm = fmaxf(lm, comp[i]);
  sv[t] = lm; __syncthreads();
  for (int s = 128; s > 0; s >>= 1) { if (t < s) sv[t] = fmaxf(sv[t], sv[t+s]); __syncthreads(); }
  if (!t) s_max = sv[0];
  __syncthreads();
  float thr = s_max * THR_;
  int lc = 0;
  for (int i = t; i < n; i += 256) {
    bool sel = comp[i] < thr;
    sc[i] = sel ? c2[i] : 0.0f;
    lc += sel ? 1 : 0;
  }
  si[t] = lc; __syncthreads();
  for (int s = 128; s > 0; s >>= 1) { if (t < s) si[t] += si[t+s]; __syncthreads(); }
  if (!t) s_cnt = (si[0] < SEED_) ? si[0] : SEED_;
  __syncthreads();
  int maxnum = s_cnt;
  for (int it = 0; it < SEED_; ++it) {
    float bv = -3.0e38f; int bi2 = n;
    for (int i = t; i < n; i += 256) {
      float v = sc[i];
      if (v > bv || (v == bv && i < bi2)) { bv = v; bi2 = i; }
    }
    sv[t] = bv; si[t] = bi2; __syncthreads();
    for (int s = 128; s > 0; s >>= 1) {
      if (t < s) {
        if (sv[t+s] > sv[t] || (sv[t+s] == sv[t] && si[t+s] < si[t])) { sv[t] = sv[t+s]; si[t] = si[t+s]; }
      }
      __syncthreads();
    }
    if (!t) { sidx[it] = si[0]; smask[it] = (it < maxnum) ? 1 : 0; sc[si[0]] = -3.0e38f; }
    __syncthreads();
  }
}

// ---------------- spot selection ----------------
__global__ void k_select_spots(const int* __restrict__ self_idx, const int* __restrict__ cross_idx,
                               const float* __restrict__ c2, const int* __restrict__ mi,
                               int* __restrict__ sp, int n) {
  int q = blockIdx.x * blockDim.x + threadIdx.x;
  if (q >= n) return;
  float nc[KNN_]; int taken[KNN_];
  for (int k = 0; k < KNN_; ++k) { nc[k] = c2[self_idx[q*KNN_ + k]]; taken[k] = 0; }
  for (int s = 0; s < SPOTS_; ++s) {
    float bv = -3.0e38f; int bp = 0;
    for (int k = 0; k < KNN_; ++k) {
      if (!taken[k] && nc[k] > bv) { bv = nc[k]; bp = k; }
    }
    taken[bp] = 1;
    int ss = self_idx[q*KNN_ + bp];
    int mt = mi[ss];
    for (int j = 0; j < SPOTK_; ++j)
      sp[q*(SPOTS_*SPOTK_) + s*SPOTK_ + j] = cross_idx[mt*KNN_ + j];
  }
}

// ---------------- gathered attention (online softmax, float4 loads) ----------------
__global__ void k_sparse_attn(const float* __restrict__ q, const float* __restrict__ kall,
                              const float* __restrict__ vall, const int* __restrict__ idx,
                              int perq, int S, const int* __restrict__ mask,
                              float* __restrict__ outb, int n) {
  int t = blockIdx.x * blockDim.x + threadIdx.x;
  if (t >= n * H_) return;
  int nq = t / H_, h = t % H_;
  const float4* qr4 = (const float4*)(q + (long)nq*D_ + h*DH_);
  float4 qv[16];
#pragma unroll
  for (int i = 0; i < 16; ++i) qv[i] = qr4[i];
  float mrun = -3.0e38f, l = 0.0f;
  float4 o[16];
#pragma unroll
  for (int i = 0; i < 16; ++i) { o[i].x = 0; o[i].y = 0; o[i].z = 0; o[i].w = 0; }
  const int* ib = idx + (perq ? (long)nq*S : 0);
  for (int s = 0; s < S; ++s) {
    int kv = ib[s];
    const float4* kr4 = (const float4*)(kall + (long)kv*D_ + h*DH_);
    float dot = 0.0f;
#pragma unroll
    for (int i = 0; i < 16; ++i) {
      float4 kk = kr4[i];
      dot += qv[i].x*kk.x + qv[i].y*kk.y + qv[i].z*kk.z + qv[i].w*kk.w;
    }
    float scv = dot * SCALE_;
    if (mask && !mask[s]) scv = -1.0e9f;
    float mn = fmaxf(mrun, scv);
    float corr = __expf(mrun - mn);
    float w = __expf(scv - mn);
    l = l * corr + w;
    const float4* vr4 = (const float4*)(vall + (long)kv*D_ + h*DH_);
#pragma unroll
    for (int i = 0; i < 16; ++i) {
      float4 vv = vr4[i];
      o[i].x = o[i].x*corr + w*vv.x;
      o[i].y = o[i].y*corr + w*vv.y;
      o[i].z = o[i].z*corr + w*vv.z;
      o[i].w = o[i].w*corr + w*vv.w;
    }
    mrun = mn;
  }
  float inv = 1.0f / l;
  float4* orow = (float4*)(outb + (long)nq*D_ + h*DH_);
#pragma unroll
  for (int i = 0; i < 16; ++i) {
    float4 ov; ov.x = o[i].x*inv; ov.y = o[i].y*inv; ov.z = o[i].z*inv; ov.w = o[i].w*inv;
    orow[i] = ov;
  }
}

// ---------------- LayerNorm over D=256 (one workgroup per row) ----------------
__global__ void k_layernorm(const float* __restrict__ src, const float* __restrict__ g,
                            const float* __restrict__ b, float* __restrict__ dst) {
  int r = blockIdx.x, t = threadIdx.x; // blockDim == D_
  __shared__ float red[256];
  float v = src[(long)r*D_ + t];
  red[t] = v; __syncthreads();
  for (int s = 128; s > 0; s >>= 1) { if (t < s) red[t] += red[t+s]; __syncthreads(); }
  float mean = red[0] / (float)D_; __syncthreads();
  float d0 = v - mean;
  red[t] = d0*d0; __syncthreads();
  for (int s = 128; s > 0; s >>= 1) { if (t < s) red[t] += red[t+s]; __syncthreads(); }
  float var = red[0] / (float)D_;
  dst[(long)r*D_ + t] = d0 * rsqrtf(var + 1e-5f) * g[t] + b[t];
}

// ---------------- host orchestration ----------------
struct Layer {
  const float *Wq,*bq,*Wk,*bk,*Wv,*bv,*Wo,*bo,*g1,*be1,*W1,*b1,*W2,*b2,*g2,*be2;
};

static Layer getLayer(void* const* din, int base) {
  Layer L;
  L.Wq  = (const float*)din[base+0];  L.bq  = (const float*)din[base+1];
  L.Wk  = (const float*)din[base+2];  L.bk  = (const float*)din[base+3];
  L.Wv  = (const float*)din[base+4];  L.bv  = (const float*)din[base+5];
  L.Wo  = (const float*)din[base+6];  L.bo  = (const float*)din[base+7];
  L.g1  = (const float*)din[base+8];  L.be1 = (const float*)din[base+9];
  L.W1  = (const float*)din[base+10]; L.b1  = (const float*)din[base+11];
  L.W2  = (const float*)din[base+12]; L.b2  = (const float*)din[base+13];
  L.g2  = (const float*)din[base+14]; L.be2 = (const float*)din[base+15];
  return L;
}

extern "C" void kernel_launch(void* const* d_in, const int* in_sizes, int n_in,
                              void* d_out, int out_size, void* d_ws, size_t ws_size,
                              hipStream_t stream) {
  (void)in_sizes; (void)n_in; (void)out_size; (void)ws_size;
  const int n = BN_;
  const float* ref_points = (const float*)d_in[0];
  const float* src_points = (const float*)d_in[1];
  const float* ref_feats  = (const float*)d_in[2];
  const float* src_feats  = (const float*)d_in[3];
  const float* Wi = (const float*)d_in[4];
  const float* bi = (const float*)d_in[5];
  // d_in[6..21] = params['lin'] -> dead code in the reference (skipped)

  // workspace arena
  float* ws = (float*)d_ws;
  size_t off = 0;
  auto alloc = [&](size_t ne) { float* p = ws + off; off += (ne + 63) & ~(size_t)63; return p; };
  float* ref_d = alloc((size_t)n*n);
  float* src_d = alloc((size_t)n*n);
  float* Sbuf  = alloc((size_t)n*n);
  float* rf    = alloc((size_t)n*D_);
  float* sf    = alloc((size_t)n*D_);
  float* qb    = alloc((size_t)n*D_);
  float* kb    = alloc((size_t)n*D_);
  float* vb    = alloc((size_t)n*D_);
  float* ab    = alloc((size_t)n*D_);
  float* t1    = alloc((size_t)n*D_);
  float* xb    = alloc((size_t)n*D_);
  float* hh    = alloc((size_t)n*DFF_);
  float* t2    = alloc((size_t)n*D_);
  float* rmax  = alloc(n); float* rsum = alloc(n);
  float* cmax  = alloc(n); float* csum = alloc(n);
  float* confR = alloc(n); float* confS = alloc(n);
  float* compR = alloc(n); float* compS = alloc(n);
  float* c2R   = alloc(n); float* c2S  = alloc(n);
  int* miR    = (int*)alloc(n);
  int* miS    = (int*)alloc(n);
  int* refIdx = (int*)alloc((size_t)n*KNN_);
  int* srcIdx = (int*)alloc((size_t)n*KNN_);
  int* sidxR  = (int*)alloc(SEED_); int* smaskR = (int*)alloc(SEED_);
  int* sidxS  = (int*)alloc(SEED_); int* smaskS = (int*)alloc(SEED_);
  int* rsp    = (int*)alloc((size_t)n*SPOTS_*SPOTK_);
  int* ssp    = (int*)alloc((size_t)n*SPOTS_*SPOTK_);

  // outputs (concatenated flat, return order)
  float* out      = (float*)d_out;
  float* out_rf   = out;
  float* out_sf   = out_rf + (size_t)n*D_;
  float* out_corr = out_sf + (size_t)n*D_;
  float* out_rc   = out_corr + (size_t)NBLOCKS_*n*n;
  float* out_sc   = out_rc + (size_t)n*NBLOCKS_;

  auto gemm = [&](const float* A, int lda, const float* B, int ldb, int tB,
                  const float* bias, const float* resid, float* C, int ldc,
                  int rows, int cols, int kd, int relu) {
    dim3 g(cols/64, rows/128);
    k_gemm<<<g, 128, 0, stream>>>(A, lda, B, ldb, tB, bias, resid, C, ldc, kd, relu);
  };

  auto sparse = [&](const float* xq, const float* xkv, const int* idx, int perq, int S,
                    const int* mask, const Layer& L, float* dst) {
    gemm(xq,  D_, L.Wq, D_, 0, L.bq, nullptr, qb, D_, n, D_, D_, 0);
    gemm(xkv, D_, L.Wk, D_, 0, L.bk, nullptr, kb, D_, n, D_, D_, 0);
    gemm(xkv, D_, L.Wv, D_, 0, L.bv, nullptr, vb, D_, n, D_, D_, 0);
    k_sparse_attn<<<(n*H_)/256, 256, 0, stream>>>(qb, kb, vb, idx, perq, S, mask, ab, n);
    gemm(ab, D_, L.Wo, D_, 0, L.bo, xq, t1, D_, n, D_, D_, 0);
    k_layernorm<<<n, 256, 0, stream>>>(t1, L.g1, L.be1, xb);
    gemm(xb, D_, L.W1, DFF_, 0, L.b1, nullptr, hh, DFF_, n, DFF_, D_, 1);
    gemm(hh, DFF_, L.W2, D_, 0, L.b2, xb, t2, D_, n, D_, DFF_, 0);
    k_layernorm<<<n, 256, 0, stream>>>(t2, L.g2, L.be2, dst);
  };

  // geometry
  dim3 pg(n/256, n);
  k_pdist<<<pg, 256, 0, stream>>>(ref_points, ref_points, ref_d, n, n);
  k_pdist<<<pg, 256, 0, stream>>>(src_points, src_points, src_d, n, n);
  k_knn<<<n/256, 256, 0, stream>>>(ref_d, n, refIdx);
  k_knn<<<n/256, 256, 0, stream>>>(src_d, n, srcIdx);

  // input projection (WMMA GEMM, K=1024)
  gemm(ref_feats, CIN_, Wi, D_, 0, bi, nullptr, rf, D_, n, D_, CIN_, 0);
  gemm(src_feats, CIN_, Wi, D_, 0, bi, nullptr, sf, D_, n, D_, CIN_, 0);

  for (int b = 0; b < NBLOCKS_; ++b) {
    float* ms = out_corr + (size_t)b*n*n;
    // S = rf @ sf^T  (2048 x 2048 x 256, bf16 WMMA)
    gemm(rf, D_, sf, D_, 1, nullptr, nullptr, Sbuf, n, n, n, D_, 0);
    k_rowstats<<<n, 256, 0, stream>>>(Sbuf, n, rmax, rsum);
    k_colstats<<<n, 256, 0, stream>>>(Sbuf, n, cmax, csum);
    k_dualsoftmax<<<(n*n)/1024, 256, 0, stream>>>(Sbuf, rmax, rsum, cmax, csum, ms, n);
    k_rowargmax<<<n, 256, 0, stream>>>(ms, n, confR, miR);
    k_colargmax<<<n, 256, 0, stream>>>(ms, n, confS, miS);
    k_compat<<<n, 256, 0, stream>>>(ref_d, src_d, miR, confR, compR, c2R, out_rc, b, n);
    k_compat<<<n, 256, 0, stream>>>(src_d, ref_d, miS, confS, compS, c2S, out_sc, b, n);
    k_seeding<<<1, 256, 0, stream>>>(compR, c2R, sidxR, smaskR, n);
    k_seeding<<<1, 256, 0, stream>>>(compS, c2S, sidxS, smaskS, n);
    if (b == NBLOCKS_ - 1) {
      k_select_spots<<<n/256, 256, 0, stream>>>(refIdx, srcIdx, c2R, miR, rsp, n);
      k_select_spots<<<n/256, 256, 0, stream>>>(srcIdx, refIdx, c2S, miS, ssp, n);
    }
    Layer Lc = getLayer(d_in, 70 + 16*b);
    sparse(rf, rf, sidxR, 0, SEED_, smaskR, Lc, rf);
    sparse(sf, sf, sidxS, 0, SEED_, smaskS, Lc, sf);
    if (b == NBLOCKS_ - 1) {
      // spot layers only live on the final block (earlier results are dead)
      Layer Ls = getLayer(d_in, 22 + 16*b);
      sparse(rf, sf, rsp, 1, SPOTS_*SPOTK_, nullptr, Ls, out_rf);
      sparse(sf, rf, ssp, 1, SPOTS_*SPOTK_, nullptr, Ls, out_sf);
    }
  }
}
